// _ESRNN_21423296872654
// MI455X (gfx1250) — matrix operand order repaired and checked by reading, not compiled, not run on hardware
//
#include <hip/hip_runtime.h>

// ---------------------------------------------------------------------------
// ESRNN for MI455X (gfx1250): wave32, WMMA f32_16x16x32_f16,
// async global->LDS staging (ASYNCcnt), packed f16 math.
// ---------------------------------------------------------------------------

typedef _Float16 v16h __attribute__((ext_vector_type(16)));
typedef _Float16 v8h  __attribute__((ext_vector_type(8)));
typedef _Float16 v4h  __attribute__((ext_vector_type(4)));
typedef float    v8f  __attribute__((ext_vector_type(8)));
typedef float    v4f  __attribute__((ext_vector_type(4)));

#define NW    1993      // N_WINDOWS
#define NS    512       // N_SERIES
#define NTIME 2048
#define INSZ  28
#define OUTSZ 28
#define HSZ   64
#define G4    256       // 4*HSZ gate dim

// workspace layout (bytes)
static constexpr size_t WS_LOGNORM = 0;                                   // NS*NTIME f32 = 4 MB
static constexpr size_t WS_X0      = 4u * 1024u * 1024u;                  // NW*NS*INSZ f16 = 57.1 MB
static constexpr size_t WS_BUFA    = WS_X0 + (size_t)NW * NS * INSZ * 2;  // NW*NS*HSZ f16 = 130.6 MB
static constexpr size_t WS_BUFB    = WS_BUFA + (size_t)NW * NS * HSZ * 2;

__device__ __forceinline__ float sigm_fast(float x) { return 1.0f / (1.0f + __expf(-x)); }
__device__ __forceinline__ float tanh_fast(float x) { return 1.0f - 2.0f / (__expf(2.0f * x) + 1.0f); }

__device__ __forceinline__ v8f wmma16x16x32(v16h a, v16h b, v8f c) {
  // (neg_a, A, neg_b, B, c_mod, C, reuse_a, reuse_b)
  return __builtin_amdgcn_wmma_f32_16x16x32_f16(false, a, false, b, (short)0, c, false, false);
}

// Pin a converted f16 fragment in VGPRs so the compiler cannot rematerialize
// the f32->f16 conversion inside the recurrence loop.
__device__ __forceinline__ void pin_frag(v16h& v) { asm volatile("" : "+v"(v)); }

// gfx1250 async DMA: global -> LDS, 8 bytes per lane, tracked by ASYNCcnt.
__device__ __forceinline__ void async_load_b64(unsigned lds_off, const void* gptr) {
  unsigned long long ga = (unsigned long long)(size_t)gptr;
  asm volatile("global_load_async_to_lds_b64 %0, %1, off"
               :: "v"(lds_off), "v"(ga)
               : "memory");
}
__device__ __forceinline__ void wait_asynccnt0() {
  asm volatile("s_wait_asynccnt 0" ::: "memory");
}

// A fragment (16xK f16, MxK): lane<16 -> row=lane, K = [kt*32+0..7, kt*32+16..23]
//                             lane>=16 -> row=lane-16, K = [kt*32+8..15, kt*32+24..31]
__device__ __forceinline__ v16h lds_a_frag(const _Float16* tile, int ld, int lane, int ktile) {
  int row = lane & 15;
  int kb  = ktile * 32 + ((lane >> 4) & 1) * 8;
  const _Float16* p = tile + row * ld + kb;
  v8h lo = *(const v8h*)(p);
  v8h hi = *(const v8h*)(p + 16);
  return __builtin_shufflevector(lo, hi, 0, 1, 2, 3, 4, 5, 6, 7, 8, 9, 10, 11, 12, 13, 14, 15);
}

// B fragment (Kx16, B[k][n] = W[n][k]): lane n = lane&15, K = kt*32 + (lane<16?0:8) + (j<8 ? j : j+8)
__device__ __forceinline__ v16h gbl_b_frag(const float* __restrict__ W, int ldk, int kmax,
                                           int n, bool nok, int ktile, int lane) {
  int k0 = ktile * 32 + ((lane >> 4) & 1) * 8;
  v16h b;
#pragma unroll
  for (int j = 0; j < 16; j++) {
    int k   = k0 + (j < 8 ? j : j + 8);
    float w = (nok && k < kmax) ? W[(size_t)n * ldk + k] : 0.0f;
    b[j]    = (_Float16)w;
  }
  return b;
}

// ---------------------------------------------------------------------------
// Kernel 1: exponential smoothing scan. One thread per series.
// Time loop unrolled x8 so the 8-deep seasonal ring has static indices
// (stays in VGPRs, no scratch).
// ---------------------------------------------------------------------------
__global__ __launch_bounds__(256) void es_kernel(
    const float* __restrict__ y, const int* __restrict__ idxs,
    const float* __restrict__ lev_sms, const float* __restrict__ seas_sms,
    const float* __restrict__ init_seas,
    float* __restrict__ levels, float* __restrict__ lognorm) {
  int s = blockIdx.x * 256 + threadIdx.x;
  if (s >= NS) return;
  int idx  = idxs[s];
  float ls = 1.0f / (1.0f + expf(-lev_sms[idx]));
  float ss = 1.0f / (1.0f + expf(-seas_sms[idx]));
  float rg[8];
#pragma unroll
  for (int j = 0; j < 7; j++) rg[j] = expf(init_seas[idx * 7 + j]);
  rg[7] = rg[0];

  const float* ys = y + (size_t)s * NTIME;
  float* lv = levels + (size_t)s * NTIME;
  float* ln = lognorm + (size_t)s * NTIME;

  float lev = ys[0] / rg[0];
  lv[0] = lev;
  ln[0] = logf(ys[0] / rg[0]);

  // step t (t>=1): reads rg[(t-1+1)&7], writes rg[(t-1)&7]
  for (int tb = 1; tb < NTIME; tb += 8) {
#pragma unroll
    for (int j = 0; j < 8; j++) {
      int t = tb + j;
      if (t < NTIME) {
        float yt   = ys[t];
        float st   = rg[(j + 1) & 7];  // seasonalities[t]
        float nl   = ls * (yt / st) + (1.0f - ls) * lev;
        float nsea = ss * (yt / nl) + (1.0f - ss) * st;
        rg[j & 7] = nsea;
        lv[t] = nl;
        ln[t] = logf(yt / st);
        lev = nl;
      }
    }
  }
}

// ---------------------------------------------------------------------------
// Kernel 2: window extraction. One thread per (w,s).
// windows_y -> d_out region 1 (f32); windows_y_hat -> X0 (f16 LSTM input).
// ---------------------------------------------------------------------------
__global__ __launch_bounds__(256) void windows_kernel(
    const float* __restrict__ lognorm, const float* __restrict__ levels,
    const float* __restrict__ noise, float* __restrict__ out_wy,
    _Float16* __restrict__ X0) {
  int id = blockIdx.x * 256 + threadIdx.x;
  if (id >= NW * NS) return;
  int w = id / NS, s = id % NS;
  float ll = logf(levels[(size_t)s * NTIME + w + INSZ]);  // lev_end
  const float* lns = lognorm + (size_t)s * NTIME + w;
  size_t base = ((size_t)w * NS + s) * INSZ;
#pragma unroll 4
  for (int i = 0; i < INSZ; i++) {
    float xh = lns[i] - ll + 0.001f * noise[base + i];
    X0[base + i] = (_Float16)xh;
    out_wy[base + i] = lns[INSZ + i] - ll;  // windows_y
  }
}

// ---------------------------------------------------------------------------
// Kernel 3: fused dilated LSTM layer (input projection + recurrence + gates).
// Grid: rate*32 workgroups. WG = (sub-phase, 16-series tile), fully independent
// recurrence lane -> no inter-WG sync. 8 waves x 32 gate-columns each.
// z(16x256) = x(16xD)@Wih^T + h(16x64)@Whh^T + b via v_wmma_f32_16x16x32_f16.
// D=64 layers stage x via global_load_async_to_lds_b64 (ASYNCcnt).
// ---------------------------------------------------------------------------
template <int D>
__global__ __launch_bounds__(256) void lstm_layer_kernel(
    const _Float16* __restrict__ xin, _Float16* __restrict__ hout,
    const float* __restrict__ Wih, const float* __restrict__ Whh,
    const float* __restrict__ bih, const float* __restrict__ bhh, int rate) {
  constexpr int KX   = (D <= 32) ? 32 : 64;  // K padded to WMMA granularity
  constexpr int KT_X = KX / 32;
  __shared__ __align__(16) _Float16 xt[16][KX];
  __shared__ __align__(16) _Float16 ht[16][HSZ];
  __shared__ __align__(16) float zb[16][G4];

  const int tid  = threadIdx.x;
  const int lane = tid & 31;
  const int wv   = tid >> 5;
  const int sub  = blockIdx.x >> 5;          // dilation phase
  const int b0   = (blockIdx.x & 31) << 4;   // series tile base
  const int nsteps = (NW + rate - 1) / rate; // Tpad / rate

  // Persistent weight B-fragments in VGPRs (f16), bias per owned column.
  v16h wih[KT_X][2];
  v16h whh[2][2];
  float bias[2];
#pragma unroll
  for (int nt = 0; nt < 2; nt++) {
    int n = wv * 32 + nt * 16 + (lane & 15);
    bias[nt] = bih[n] + bhh[n];
#pragma unroll
    for (int kt = 0; kt < KT_X; kt++) {
      wih[kt][nt] = gbl_b_frag(Wih, D, D, n, true, kt, lane);
      pin_frag(wih[kt][nt]);
    }
#pragma unroll
    for (int kt = 0; kt < 2; kt++) {
      whh[kt][nt] = gbl_b_frag(Whh, HSZ, HSZ, n, true, kt, lane);
      pin_frag(whh[kt][nt]);
    }
  }

  // Gate-phase ownership: thread -> (row gm, 4 consecutive units gu..gu+3).
  const int gm = tid >> 4;
  const int gu = (tid & 15) * 4;
  v4f cst = {0.f, 0.f, 0.f, 0.f};  // cell state (registers, fixed mapping)
  {
    _Float16* hp = &ht[0][0];
    for (int e = tid; e < 16 * HSZ; e += 256) hp[e] = (_Float16)0.f;
  }
  __syncthreads();

  for (int td = 0; td < nsteps; td++) {
    const int t = td * rate + sub;
    const bool valid = (t < NW);  // zero padding beyond T (uniform per block)

    // ---- stage x tile into LDS (f16) ----
    if (D == 64) {
      if (valid) {
        // async DMA: 256 threads x 8B = full 16x64 f16 tile
        int m = tid >> 4, k = (tid & 15) * 4;
        unsigned lds_off = (unsigned)(size_t)&xt[m][k];
        const _Float16* g = xin + ((size_t)t * NS + b0 + m) * (size_t)D + k;
        async_load_b64(lds_off, g);
        wait_asynccnt0();
      } else {
        _Float16* xp = &xt[0][0];
        for (int e = tid; e < 16 * KX; e += 256) xp[e] = (_Float16)0.f;
      }
    } else {
      _Float16* xp = &xt[0][0];
      for (int e = tid; e < 16 * KX; e += 256) {
        int m = e / KX, k = e % KX;
        _Float16 v = (_Float16)0.f;
        if (valid && k < D) v = xin[((size_t)t * NS + b0 + m) * D + k];
        xp[e] = v;
      }
    }
    if (valid && (t + rate) < NW)
      __builtin_prefetch(xin + ((size_t)(t + rate) * NS + b0) * D, 0, 1);
    __syncthreads();

    // ---- z tile: 2 N-tiles per wave, K-chained WMMAs ----
    v8f acc0 = {0.f, 0.f, 0.f, 0.f, 0.f, 0.f, 0.f, 0.f};
    v8f acc1 = acc0;
#pragma unroll
    for (int kt = 0; kt < KT_X; kt++) {
      v16h a = lds_a_frag(&xt[0][0], KX, lane, kt);
      acc0 = wmma16x16x32(a, wih[kt][0], acc0);
      acc1 = wmma16x16x32(a, wih[kt][1], acc1);
    }
#pragma unroll
    for (int kt = 0; kt < 2; kt++) {
      v16h a = lds_a_frag(&ht[0][0], HSZ, lane, kt);
      acc0 = wmma16x16x32(a, whh[kt][0], acc0);
      acc1 = wmma16x16x32(a, whh[kt][1], acc1);
    }

    // spill z to LDS with bias (D layout: VGPR j -> row j or j+8, col = lane&15)
    {
      int mb = ((lane >> 4) & 1) * 8;
      int nc = lane & 15;
#pragma unroll
      for (int j = 0; j < 8; j++) {
        zb[mb + j][wv * 32 + nc]      = acc0[j] + bias[0];
        zb[mb + j][wv * 32 + 16 + nc] = acc1[j] + bias[1];
      }
    }
    __syncthreads();

    // ---- gates: i,f,g,o blocks of 64; vectorized 4 units/thread ----
    {
      v4f zi = *(const v4f*)&zb[gm][gu];
      v4f zf = *(const v4f*)&zb[gm][HSZ + gu];
      v4f zg = *(const v4f*)&zb[gm][2 * HSZ + gu];
      v4f zo = *(const v4f*)&zb[gm][3 * HSZ + gu];
      v4h h4;
#pragma unroll
      for (int q = 0; q < 4; q++) {
        float cn = sigm_fast(zf[q]) * cst[q] + sigm_fast(zi[q]) * tanh_fast(zg[q]);
        float hn = sigm_fast(zo[q]) * tanh_fast(cn);
        cst[q] = cn;
        h4[q]  = (_Float16)hn;
      }
      *(v4h*)&ht[gm][gu] = h4;
      if (valid) *(v4h*)(hout + ((size_t)t * NS + b0 + gm) * HSZ + gu) = h4;
    }
    __syncthreads();
  }
}

// ---------------------------------------------------------------------------
// Kernel 4: fused residual add + output projection (x = h3 + res) @ adW^T + adb
// One 16-row WMMA tile per wave; N=28 padded to 2 N-tiles.
// ---------------------------------------------------------------------------
__global__ __launch_bounds__(256) void proj_kernel(
    const _Float16* __restrict__ hA, const _Float16* __restrict__ hB,
    const float* __restrict__ adW, const float* __restrict__ adb,
    float* __restrict__ out) {
  __shared__ __align__(16) _Float16 xt[8][16][HSZ];
  const int tid = threadIdx.x, lane = tid & 31, wv = tid >> 5;
  const size_t tile = (size_t)blockIdx.x * 8 + wv;
  const size_t row0 = tile * 16;

  v16h wfrag[2][2];
  float bn[2];
#pragma unroll
  for (int nt = 0; nt < 2; nt++) {
    int n = nt * 16 + (lane & 15);
    bool nok = (n < OUTSZ);
    bn[nt] = nok ? adb[n] : 0.f;
#pragma unroll
    for (int kt = 0; kt < 2; kt++) {
      wfrag[kt][nt] = gbl_b_frag(adW, HSZ, HSZ, n, nok, kt, lane);
      pin_frag(wfrag[kt][nt]);
    }
  }

  {  // residual add in packed f16 (v_pk_add_f16), 32 halves per lane
    int m = lane >> 1;
    int k0 = (lane & 1) * 32;
#pragma unroll
    for (int c = 0; c < 4; c++) {
      size_t off = (row0 + m) * HSZ + k0 + c * 8;
      v8h a = *(const v8h*)(hA + off);
      v8h b = *(const v8h*)(hB + off);
      *(v8h*)&xt[wv][m][k0 + c * 8] = a + b;
    }
  }
  __syncthreads();

  v8f acc0 = {0.f, 0.f, 0.f, 0.f, 0.f, 0.f, 0.f, 0.f};
  v8f acc1 = acc0;
#pragma unroll
  for (int kt = 0; kt < 2; kt++) {
    v16h a = lds_a_frag(&xt[wv][0][0], HSZ, lane, kt);
    acc0 = wmma16x16x32(a, wfrag[kt][0], acc0);
    acc1 = wmma16x16x32(a, wfrag[kt][1], acc1);
  }

  int mb = ((lane >> 4) & 1) * 8;
  int nc = lane & 15;
#pragma unroll
  for (int j = 0; j < 8; j++) {
    size_t r = row0 + mb + j;
    out[r * OUTSZ + nc] = acc0[j] + bn[0];  // n = nc < 28 always
    if (16 + nc < OUTSZ) out[r * OUTSZ + 16 + nc] = acc1[j] + bn[1];
  }
}

// ---------------------------------------------------------------------------
extern "C" void kernel_launch(void* const* d_in, const int* in_sizes, int n_in,
                              void* d_out, int out_size, void* d_ws, size_t ws_size,
                              hipStream_t stream) {
  (void)in_sizes; (void)n_in; (void)out_size; (void)ws_size;

  const float* y         = (const float*)d_in[0];
  const int*   idxs      = (const int*)d_in[1];
  const float* noise     = (const float*)d_in[2];
  const float* lev_sms   = (const float*)d_in[3];
  const float* seas_sms  = (const float*)d_in[4];
  const float* init_seas = (const float*)d_in[5];
  const float* Wih0      = (const float*)d_in[6];
  const float* Wih_rest  = (const float*)d_in[7];
  const float* Whh_all   = (const float*)d_in[8];
  const float* bih_all   = (const float*)d_in[9];
  const float* bhh_all   = (const float*)d_in[10];
  const float* adW       = (const float*)d_in[11];
  const float* adb       = (const float*)d_in[12];

  float* out      = (float*)d_out;
  float* out_wy   = out;                                // windows_y       (NW,NS,28)
  float* out_yhat = out + (size_t)NW * NS * OUTSZ;      // windows_y_hat_out
  float* out_lev  = out + 2 * (size_t)NW * NS * OUTSZ;  // levels (NS,NTIME)

  char* ws = (char*)d_ws;
  float*    lognorm = (float*)(ws + WS_LOGNORM);
  _Float16* X0      = (_Float16*)(ws + WS_X0);
  _Float16* bufA    = (_Float16*)(ws + WS_BUFA);
  _Float16* bufB    = (_Float16*)(ws + WS_BUFB);

  es_kernel<<<2, 256, 0, stream>>>(y, idxs, lev_sms, seas_sms, init_seas, out_lev, lognorm);
  windows_kernel<<<(NW * NS + 255) / 256, 256, 0, stream>>>(lognorm, out_lev, noise, out_wy, X0);

  // group 0: rates 1,2 (layers 0,1);  group 1: rates 4,8 (layers 2,3) + residual
  lstm_layer_kernel<28><<<1 * 32, 256, 0, stream>>>(X0,   bufA, Wih0,                    Whh_all + 0 * G4 * HSZ, bih_all + 0 * G4, bhh_all + 0 * G4, 1);
  lstm_layer_kernel<64><<<2 * 32, 256, 0, stream>>>(bufA, bufB, Wih_rest + 0 * G4 * HSZ, Whh_all + 1 * G4 * HSZ, bih_all + 1 * G4, bhh_all + 1 * G4, 2);
  lstm_layer_kernel<64><<<4 * 32, 256, 0, stream>>>(bufB, bufA, Wih_rest + 1 * G4 * HSZ, Whh_all + 2 * G4 * HSZ, bih_all + 2 * G4, bhh_all + 2 * G4, 4);
  // layer 3 in-place on bufA (each WG reads x[t] before writing h[t]; disjoint rows)
  lstm_layer_kernel<64><<<8 * 32, 256, 0, stream>>>(bufA, bufA, Wih_rest + 2 * G4 * HSZ, Whh_all + 3 * G4 * HSZ, bih_all + 3 * G4, bhh_all + 3 * G4, 8);

  // out = (h3 + residual(bufB)) @ adW^T + adb ; rows = NW*NS = 63776 tiles, 8/block
  proj_kernel<<<(NW * NS / 16) / 8, 256, 0, stream>>>(bufA, bufB, adW, adb, out_yhat);
}
